// AttentionMechanism_30562987278373
// MI455X (gfx1250) — matrix operand order repaired and checked by reading, not compiled, not run on hardware
//
#include <hip/hip_runtime.h>

// ---------------------------------------------------------------------------
// Fused KD-attention kernel for MI455X (gfx1250, wave32, WMMA).
//   t_proj = t_output @ w1^T + b1 ; s_proj = s_output @ w2^T + b2
//   kl[t,n] = sum_h softmax(t_proj)*(log_softmax(t_proj)-log_softmax(s_proj))
//   w = softmax_t(-kl/sqrt(128)) ; y = sum_t w[t,n]*t_output[t,n,:]
// One wave == one 16-node tile. GEMMs via v_wmma_f32_16x16x32_f16.
// Teacher loop kept rolled for occupancy; A-fragments double-buffered from
// an s-major LDS image so ds_load latency overlaps WMMA issue.
// ---------------------------------------------------------------------------

typedef _Float16 v16h __attribute__((ext_vector_type(16)));
typedef float    v8f  __attribute__((ext_vector_type(8)));

#define T_MODELS 8
#define LOG2E 1.4426950408889634f
#define LN2   0.6931471805599453f

union AFrag { uint4 q[2]; v16h h; };

// Build the 4 B-fragments (32ch x 16nodes each, f16) for one activation row.
// B k order used here (contiguous, not AMD's table): lanes 0-15 hold K=0..15, lanes 16-31 hold K=16..31,
// VGPR v = K pair (2v, 2v+1)  ->  lane needs 16 *contiguous* channels:
//   channels [s*32 + 16*half .. +15]
__device__ __forceinline__ void load_act_f16(const float* __restrict__ row,
                                             int half, v16h B[4]) {
#pragma unroll
  for (int s = 0; s < 4; ++s) {
    const float4* p = (const float4*)(row + s * 32 + half * 16);
    float4 f0 = p[0], f1 = p[1], f2 = p[2], f3 = p[3];
    v16h b;
    b[0]  = (_Float16)f0.x; b[1]  = (_Float16)f0.y;
    b[2]  = (_Float16)f0.z; b[3]  = (_Float16)f0.w;
    b[4]  = (_Float16)f1.x; b[5]  = (_Float16)f1.y;
    b[6]  = (_Float16)f1.z; b[7]  = (_Float16)f1.w;
    b[8]  = (_Float16)f2.x; b[9]  = (_Float16)f2.y;
    b[10] = (_Float16)f2.z; b[11] = (_Float16)f2.w;
    b[12] = (_Float16)f3.x; b[13] = (_Float16)f3.y;
    b[14] = (_Float16)f3.z; b[15] = (_Float16)f3.w;
    B[s] = b;
  }
}

// Full 128x128 projection of a 16-node tile: acc[hb] = 16x16 fp32 D tile,
// D[m=h, n=node]. C initialized with the bias (so bias comes for free).
// A-fragment image in LDS is s-major: frag(s,hb) at word (s*8+hb)*256+lane*8.
// s-outer / hb-inner: consecutive WMMAs hit independent accumulators and
// share B[s]; the A-fragment for each WMMA was ds_load'ed one step earlier.
__device__ __forceinline__ void proj_tile(const unsigned* __restrict__ img,
                                          const float* __restrict__ bias,
                                          const v16h B[4], int lane, int half,
                                          v8f acc[8]) {
#pragma unroll
  for (int hb = 0; hb < 8; ++hb) {
    const float4* bp = (const float4*)(bias + hb * 16 + half * 8);
    float4 c0 = bp[0], c1 = bp[1];
    v8f c;
    c[0] = c0.x; c[1] = c0.y; c[2] = c0.z; c[3] = c0.w;
    c[4] = c1.x; c[5] = c1.y; c[6] = c1.z; c[7] = c1.w;
    acc[hb] = c;
  }
  const uint4* base = (const uint4*)(img + lane * 8);  // frag idx stride: 64
  AFrag cur, nxt;
  cur.q[0] = base[0];
  cur.q[1] = base[1];
#pragma unroll
  for (int s = 0; s < 4; ++s) {
#pragma unroll
    for (int hb = 0; hb < 8; ++hb) {
      const int idx = s * 8 + hb;
      if (idx < 31) {                       // prefetch next A-fragment
        const uint4* np = base + (idx + 1) * 64;
        nxt.q[0] = np[0];
        nxt.q[1] = np[1];
      }
      acc[hb] = __builtin_amdgcn_wmma_f32_16x16x32_f16(
                    /*neg_a=*/false, cur.h, /*neg_b=*/false, B[s],
                    /*c_mod=*/(short)0, acc[hb],
                    /*reuse_a=*/false, /*reuse_b=*/false);
      cur = nxt;
    }
  }
}

// log-sum-exp over H=128 for this lane's node column: 64 local values
// (rows hb*16 + v + 8*half) + one SWAPX16 exchange with the partner lane.
__device__ __forceinline__ void lse_col(const v8f acc[8], float& m, float& sum) {
  m = -3.402823466e38f;
#pragma unroll
  for (int hb = 0; hb < 8; ++hb)
#pragma unroll
    for (int v = 0; v < 8; ++v) m = fmaxf(m, acc[hb][v]);
  m = fmaxf(m, __shfl_xor(m, 16, 32));
  sum = 0.f;
#pragma unroll
  for (int hb = 0; hb < 8; ++hb)
#pragma unroll
    for (int v = 0; v < 8; ++v) sum += exp2f((acc[hb][v] - m) * LOG2E);
  sum += __shfl_xor(sum, 16, 32);
}

__global__ void __launch_bounds__(256)
AttentionMechanism_30562987278373_kernel(
    const float* __restrict__ s_out, const float* __restrict__ t_out,
    const float* __restrict__ w1w, const float* __restrict__ w1b,
    const float* __restrict__ w2w, const float* __restrict__ w2b,
    float* __restrict__ out, int N, int ntiles) {
  extern __shared__ unsigned char smem[];
  unsigned* w1img = (unsigned*)smem;         // 8192 words (32 KB) f16 A-image
  unsigned* w2img = w1img + 8192;            // 8192 words (32 KB)
  float* b1 = (float*)(w2img + 8192);        // 128 floats
  float* b2 = b1 + 128;                      // 128 floats
  float* klbuf = b2 + 128;                   // 8 * 256 floats (8 KB) scratch

  // --- one-time per-WG: swizzle fp32 weights into f16 A-fragment images ---
  // Image is s-major: mat = s*8 + hb.
  // A (16-bit 16x32) per-lane K map: VGPR v -> K = 2v + 16*half (matches B)
  for (int idx = threadIdx.x; idx < 16384; idx += 256) {
    int wi = idx >> 13, k = idx & 8191;
    int mat = k >> 8, ln = (k >> 3) & 31, v = k & 7;
    int s = mat >> 3, hb = mat & 7;
    int row = hb * 16 + (ln & 15), hf = ln >> 4;
    int klo = s * 32 + hf * 16 + 2 * v;
    const float* W = wi ? w2w : w1w;
    float f0 = W[row * 128 + klo], f1 = W[row * 128 + klo + 1];
    union { _Float16 h[2]; unsigned u; } pk;
    pk.h[0] = (_Float16)f0;
    pk.h[1] = (_Float16)f1;
    (wi ? w2img : w1img)[k] = pk.u;
  }
  if (threadIdx.x < 128) {
    b1[threadIdx.x] = w1b[threadIdx.x];
    b2[threadIdx.x] = w2b[threadIdx.x];
  }
  __syncthreads();

  const int wave = threadIdx.x >> 5, lane = threadIdx.x & 31;
  const long long tile = (long long)blockIdx.x * 8 + wave;
  if (tile >= ntiles) return;                 // wave-uniform: EXEC stays full
  const int col = lane & 15, half = lane >> 4;
  const long long node = tile * 16 + col;
  const long long nodec = node < N ? node : (long long)N - 1;

  v16h B[4];
  v8f acc[8];

  // ---- student: ls = log_softmax(s_proj) kept in registers ----
  load_act_f16(s_out + nodec * 128, half, B);
  proj_tile(w2img, b2, B, lane, half, acc);
  float ms_, ss_;
  lse_col(acc, ms_, ss_);
  const float Ls = ms_ + log2f(ss_) * LN2;
  float lsv[8][8];
#pragma unroll
  for (int hb = 0; hb < 8; ++hb)
#pragma unroll
    for (int v = 0; v < 8; ++v) lsv[hb][v] = acc[hb][v] - Ls;

  // ---- teachers (ROLLED loop: one iteration in flight -> low VGPR count,
  //      high occupancy). kl[t] parked in per-thread LDS scratch. ----
#pragma unroll 1
  for (int t = 0; t < T_MODELS; ++t) {
    load_act_f16(t_out + ((long long)t * N + nodec) * 128, half, B);
    if (t + 1 < T_MODELS)
      __builtin_prefetch(t_out + ((long long)(t + 1) * N + nodec) * 128, 0, 3);
    proj_tile(w1img, b1, B, lane, half, acc);
    float mt, st;
    lse_col(acc, mt, st);
    const float Lt = mt + log2f(st) * LN2;
    float part = 0.f;
#pragma unroll
    for (int hb = 0; hb < 8; ++hb)
#pragma unroll
      for (int v = 0; v < 8; ++v) {
        float x = acc[hb][v];
        float e = exp2f((x - mt) * LOG2E);        // e/st == p_t
        part = fmaf(e, (x - Lt) - lsv[hb][v], part);
      }
    part += __shfl_xor(part, 16, 32);
    klbuf[t * 256 + threadIdx.x] = part / st;     // private slot, no barrier
  }

  // ---- softmax over the 8 teachers (per node; lanes n and n+16 identical) ---
  const float sc_scale = -0.08838834764831845f;   // -1/sqrt(128), TEMP=1
  float w[T_MODELS], mx = -3.402823466e38f;
#pragma unroll
  for (int t = 0; t < T_MODELS; ++t) {
    w[t] = klbuf[t * 256 + threadIdx.x] * sc_scale;
    mx = fmaxf(mx, w[t]);
  }
  float se = 0.f;
#pragma unroll
  for (int t = 0; t < T_MODELS; ++t) {
    w[t] = exp2f((w[t] - mx) * LOG2E);
    se += w[t];
  }
  const float rs = 1.f / se;

  // ---- y[node, :] = sum_t w[t] * t_output[t, node, :] ----
  // lane handles channels [half*64 .. +63] of its node; re-reads hit WGP$/L2.
  float4 y[16];
#pragma unroll
  for (int j = 0; j < 16; ++j) y[j] = make_float4(0.f, 0.f, 0.f, 0.f);
  const int coff = half * 64;
#pragma unroll 1
  for (int t = 0; t < T_MODELS; ++t) {
    const float4* p =
        (const float4*)(t_out + ((long long)t * N + nodec) * 128 + coff);
    const float wt = w[t] * rs;
#pragma unroll
    for (int j = 0; j < 16; ++j) {
      float4 v = p[j];
      y[j].x = fmaf(wt, v.x, y[j].x);
      y[j].y = fmaf(wt, v.y, y[j].y);
      y[j].z = fmaf(wt, v.z, y[j].z);
      y[j].w = fmaf(wt, v.w, y[j].w);
    }
  }
  if (node < N) {                               // divergence only at store
    float4* o = (float4*)(out + node * 128 + coff);
#pragma unroll
    for (int j = 0; j < 16; ++j) o[j] = y[j];
  }
}

extern "C" void kernel_launch(void* const* d_in, const int* in_sizes, int n_in,
                              void* d_out, int out_size, void* d_ws,
                              size_t ws_size, hipStream_t stream) {
  const float* s_out = (const float*)d_in[0];   // [N,128]
  const float* t_out = (const float*)d_in[1];   // [8,N,128]
  const float* w1w = (const float*)d_in[2];     // [128,128]
  const float* w1b = (const float*)d_in[3];     // [128]
  const float* w2w = (const float*)d_in[4];     // [128,128]
  const float* w2b = (const float*)d_in[5];     // [128]
  float* out = (float*)d_out;                   // [N,128]

  const int N = in_sizes[0] / 128;
  const int ntiles = (N + 15) / 16;             // N=100000 -> exactly 6250
  const int blocks = (ntiles + 7) / 8;          // 8 waves (tiles) per WG
  // 64 KB weight images + 1 KB biases + 8 KB kl scratch
  const size_t shmem = (8192 + 8192) * 4 + 256 * 4 + 2048 * 4;

  hipLaunchKernelGGL(AttentionMechanism_30562987278373_kernel, dim3(blocks),
                     dim3(256), shmem, stream, s_out, t_out, w1w, w1b, w2w,
                     w2b, out, N, ntiles);
}
